// CRFNet_12756052869369
// MI455X (gfx1250) — matrix operand order repaired and checked by reading, not compiled, not run on hardware
//
#include <hip/hip_runtime.h>
#include <hip/hip_bf16.h>
#include <stdint.h>

#define T_LEN 2048
#define E_DIM 256
#define HD 256
#define G4 1024   // 4*HD
#define H_DIM 512
#define K_TAGS 12
#define START_TAG 10
#define STOP_TAG 11
#define NEG_VAL -10000.0f

typedef __attribute__((ext_vector_type(16))) __bf16    v16bf;
typedef __attribute__((ext_vector_type(8)))  float     v8f;
typedef __attribute__((ext_vector_type(8)))  unsigned  v8u;

// ---- bf16 helpers (RNE), carried as packed uint pairs -----------------------
__device__ inline unsigned short f2bf(float f) {
  unsigned u = __builtin_bit_cast(unsigned, f);
  u += 0x7FFFu + ((u >> 16) & 1u);
  return (unsigned short)(u >> 16);
}
__device__ inline unsigned pack2(float lo, float hi) {
  return (unsigned)f2bf(lo) | ((unsigned)f2bf(hi) << 16);
}
__device__ inline float bflo(unsigned u){ return __builtin_bit_cast(float, u << 16); }
__device__ inline float bfhi(unsigned u){ return __builtin_bit_cast(float, u & 0xffff0000u); }

// ---- 1) Prep: embedding gather + bf16 pair-packed weight layouts ------------
// XP    [T][E/2]          : X bf16 pairs (consecutive e)
// WihP  [dir][E/2][G4]    : B-layout pairs, pair p = (e=2p,2p+1), col j
// WhhTP [dir][HD/2][G4]   : same layout for recurrent weights
// biasC [dir][G4]         : bih + bhh
// WfcP  [H/2][16]         : Wfc^T bf16 pairs, cols >= K zeroed
__global__ void prep_kernel(const int* __restrict__ sent, const float* __restrict__ emb,
                            const float* __restrict__ WihF, const float* __restrict__ WihB,
                            const float* __restrict__ WhhF, const float* __restrict__ WhhB,
                            const float* __restrict__ bihF, const float* __restrict__ bhhF,
                            const float* __restrict__ bihB, const float* __restrict__ bhhB,
                            const float* __restrict__ Wfc,
                            unsigned* __restrict__ XP, unsigned* __restrict__ WihP,
                            unsigned* __restrict__ WhhTP, float* __restrict__ biasC,
                            unsigned* __restrict__ WfcP)
{
  const long long NX = (long long)T_LEN * 128;
  const long long NW = 128LL * G4;
  const long long TOTAL = NX + 2*NW + 2*NW + 2*G4 + 256*16;
  for (long long gid = blockIdx.x * (long long)blockDim.x + threadIdx.x;
       gid < TOTAL; gid += (long long)gridDim.x * blockDim.x) {
    long long id = gid;
    if (id < NX) {
      int t = (int)(id >> 7), p = (int)(id & 127);
      const float* e = emb + (long long)sent[t] * E_DIM + 2*p;
      XP[id] = pack2(e[0], e[1]);
      continue;
    }
    id -= NX;
    if (id < 2*NW) {
      int d = (int)(id / NW); long long r = id % NW;
      int p = (int)(r >> 10), j = (int)(r & 1023);
      const float* W = d ? WihB : WihF;
      WihP[id] = pack2(W[(long long)j*E_DIM + 2*p], W[(long long)j*E_DIM + 2*p + 1]);
      continue;
    }
    id -= 2*NW;
    if (id < 2*NW) {
      int d = (int)(id / NW); long long r = id % NW;
      int p = (int)(r >> 10), j = (int)(r & 1023);
      const float* W = d ? WhhB : WhhF;
      WhhTP[id] = pack2(W[(long long)j*HD + 2*p], W[(long long)j*HD + 2*p + 1]);
      continue;
    }
    id -= 2*NW;
    if (id < 2*G4) {
      int d = (int)(id >> 10), j = (int)(id & 1023);
      biasC[id] = d ? (bihB[j] + bhhB[j]) : (bihF[j] + bhhF[j]);
      continue;
    }
    id -= 2*G4;
    {
      int p = (int)(id >> 4), k = (int)(id & 15);
      WfcP[id] = (k < K_TAGS)
        ? pack2(Wfc[(long long)k*H_DIM + 2*p], Wfc[(long long)k*H_DIM + 2*p + 1]) : 0u;
    }
  }
}

// ---- 2) Input-projection GEMM: GX[d][t][j] = X@Wih^T + (bih+bhh) ------------
// Block = 256 threads (8 waves). Each block: 16 rows x 128 cols; wave w owns a
// 16x16 C tile (cols w*16..). K=256 in 8 unrolled steps of 32, LDS-staged.
__global__ __launch_bounds__(256)
void inproj_gemm(const unsigned* __restrict__ XP, const unsigned* __restrict__ WihP,
                 const float* __restrict__ biasC, float* __restrict__ GX)
{
  __shared__ unsigned Abuf[16*16];    // 16 rows x 16 k-pairs
  __shared__ unsigned Bbuf[16*128];   // 16 k-pairs x 128 cols
  const int tid  = threadIdx.x;
  const int lane = tid & 31;
  const int wave = tid >> 5;
  const int mtile = blockIdx.x;       // 0..127  (t tiles)
  const int nblk  = blockIdx.y;       // 0..7    (128-col blocks)
  const int dir   = blockIdx.z;       // 0..1
  const unsigned* Wp = WihP + (long long)dir * (128LL*G4);
  const int t0 = mtile * 16;
  const int j0 = nblk * 128;

  const int half = lane >> 4;         // half-wave selects K sub-range
  const int mn   = lane & 15;
  const int pA_base = half * 4;       // A: +4 pairs (k+8) for lanes 16..31
  const int pB_base = half * 4;       // B: +4 pairs (k+8) for lanes 16..31

  v8f acc = {0.f,0.f,0.f,0.f,0.f,0.f,0.f,0.f};

  #pragma unroll
  for (int kk = 0; kk < 8; ++kk) {
    const int kp0 = kk * 16;          // pair offset of this 32-wide K step
    { // stage A: 256 uints, one per thread
      int m = tid >> 4, p = tid & 15;
      Abuf[tid] = XP[(long long)(t0 + m)*128 + kp0 + p];
    }
    #pragma unroll
    for (int i = 0; i < 8; ++i) {     // stage B: 2048 uints, 8 per thread
      int idx = tid + i*256;
      int pr = idx >> 7, col = idx & 127;
      Bbuf[idx] = Wp[(long long)(kp0 + pr)*G4 + j0 + col];
      if (i == 0 && kk < 7)           // hint next K-step into cache
        __builtin_prefetch(&Wp[(long long)(kp0 + 16 + pr)*G4 + j0 + col], 0, 1);
    }
    __syncthreads();
    v8u au, bu;
    #pragma unroll
    for (int v = 0; v < 8; ++v) {
      int pa = (v < 4 ? v : v + 4) + pA_base;      // k pairs 0..7 / 8..15
      au[v] = Abuf[mn*16 + pa];
      bu[v] = Bbuf[((v < 4 ? v : v + 4) + pB_base)*128 + (wave*16 + mn)];
    }
    v16bf a = __builtin_bit_cast(v16bf, au);
    v16bf b = __builtin_bit_cast(v16bf, bu);
    acc = __builtin_amdgcn_wmma_f32_16x16x32_bf16(false, a, false, b,
                                                  (short)0, acc, false, false);
    __syncthreads();
  }
  const int j = j0 + wave*16 + mn;
  const float bias = biasC[dir*G4 + j];
  float* out = GX + ((long long)dir*T_LEN + t0)*G4 + j;
  #pragma unroll
  for (int v = 0; v < 8; ++v) {
    int m = v + half*8;
    out[(long long)m*G4] = acc[v] + bias;
  }
}

// ---- 3) LSTM recurrence (serial over T; one WGP-filling block per dir) ------
// Matvec h@Whh^T per step: thread j owns gate j; coalesced bf16-pair loads of
// WhhTP (L2-resident), h broadcast from LDS as bf16 pairs.
__global__ __launch_bounds__(1024)
void lstm_kernel(const unsigned* __restrict__ WhhTP, const float* __restrict__ GX,
                 unsigned* __restrict__ hsP)
{
  __shared__ float    g_sh[G4];
  __shared__ float    h_sh[HD];
  __shared__ unsigned hP[128];
  const int j   = threadIdx.x;
  const int dir = blockIdx.x;
  const unsigned* W  = WhhTP + (long long)dir * (128LL*G4);
  const float* gxb   = GX + (long long)dir * T_LEN * G4;
  if (j < 128) hP[j] = 0u;
  float c = 0.f;
  __syncthreads();
  for (int step = 0; step < T_LEN; ++step) {
    const int t = dir ? (T_LEN - 1 - step) : step;
    float acc = 0.f;
    #pragma unroll 8
    for (int p = 0; p < 128; ++p) {
      unsigned wv = W[(long long)p*G4 + j];
      unsigned hv = hP[p];
      acc = fmaf(bflo(wv), bflo(hv), acc);
      acc = fmaf(bfhi(wv), bfhi(hv), acc);
    }
    const float g = acc + gxb[(long long)t*G4 + j];
    float a;
    if (j >= 2*HD && j < 3*HD) {              // cell gate: tanh
      float e = __expf(2.f*g);
      a = (e - 1.f) / (e + 1.f);
    } else {                                  // i/f/o: sigmoid
      a = 1.f / (1.f + __expf(-g));
    }
    g_sh[j] = a;
    __syncthreads();
    if (j < HD) {
      c = g_sh[HD + j]*c + g_sh[j]*g_sh[2*HD + j];
      float e  = __expf(2.f*c);
      float th = (e - 1.f) / (e + 1.f);
      h_sh[j] = g_sh[3*HD + j] * th;
    }
    __syncthreads();
    if (j < 128) {
      unsigned u = pack2(h_sh[2*j], h_sh[2*j + 1]);
      hP[j] = u;                                      // feed next step
      hsP[(long long)t*256 + dir*128 + j] = u;        // concat layout [t][512]
    }
    __syncthreads();
  }
}

// ---- 4) FC feats: [T,512]@Wfc^T + bfc via WMMA (K padded 12->16) ------------
__global__ __launch_bounds__(32)
void fc_feats(const unsigned* __restrict__ hsP, const unsigned* __restrict__ WfcP,
              const float* __restrict__ bfc, float* __restrict__ feats)
{
  const int lane = threadIdx.x & 31;
  const int t0 = blockIdx.x * 16;
  const int half = lane >> 4, mn = lane & 15;
  const int pA_base = half*4, pB_base = half*4;
  v8f acc = {0.f,0.f,0.f,0.f,0.f,0.f,0.f,0.f};
  #pragma unroll
  for (int kk = 0; kk < 16; ++kk) {
    const int kp0 = kk * 16;
    v8u au, bu;
    #pragma unroll
    for (int v = 0; v < 8; ++v) {
      int pa = (v < 4 ? v : v + 4) + pA_base;
      au[v] = hsP[(long long)(t0 + mn)*256 + kp0 + pa];
      bu[v] = WfcP[(long long)(kp0 + pB_base + (v < 4 ? v : v + 4))*16 + mn];
    }
    v16bf a = __builtin_bit_cast(v16bf, au);
    v16bf b = __builtin_bit_cast(v16bf, bu);
    acc = __builtin_amdgcn_wmma_f32_16x16x32_bf16(false, a, false, b,
                                                  (short)0, acc, false, false);
  }
  if (mn < K_TAGS) {
    float bb = bfc[mn];
    #pragma unroll
    for (int v = 0; v < 8; ++v) {
      int m = v + half*8;
      feats[(long long)(t0 + m)*K_TAGS + mn] = acc[v] + bb;
    }
  }
}

// ---- 5) Viterbi: one wave, K=12; lane 0 backtracks --------------------------
__global__ __launch_bounds__(32)
void viterbi_kernel(const float* __restrict__ feats, const float* __restrict__ trans,
                    int* __restrict__ bp, float* __restrict__ out)
{
  __shared__ float fv[16];
  __shared__ float term[16];
  const int lane = threadIdx.x;
  float trow[K_TAGS];
  if (lane < K_TAGS) {
    #pragma unroll
    for (int p = 0; p < K_TAGS; ++p) trow[p] = trans[lane*K_TAGS + p];
    fv[lane] = (lane == START_TAG) ? 0.f : NEG_VAL;
  }
  __syncthreads();
  for (int t = 0; t < T_LEN; ++t) {
    float nf = 0.f;
    if (lane < K_TAGS) {
      float best = -3.4e38f; int bi = 0;
      #pragma unroll
      for (int p = 0; p < K_TAGS; ++p) {
        float s = fv[p] + trow[p];
        if (s > best) { best = s; bi = p; }
      }
      nf = best + feats[(long long)t*K_TAGS + lane];
      bp[(long long)t*K_TAGS + lane] = bi;
    }
    __syncthreads();
    if (lane < K_TAGS) fv[lane] = nf;
    __syncthreads();
  }
  if (lane < K_TAGS) term[lane] = fv[lane] + trans[STOP_TAG*K_TAGS + lane];
  __syncthreads();
  if (lane == 0) {
    float best = term[0]; int bi = 0;
    for (int k = 1; k < K_TAGS; ++k) if (term[k] > best) { best = term[k]; bi = k; }
    out[0] = best;
    int tag = bi;
    for (int t = T_LEN - 1; t >= 0; --t) {
      out[1 + t] = (float)tag;
      tag = bp[(long long)t*K_TAGS + tag];
    }
  }
}

// ---- Host launcher ----------------------------------------------------------
extern "C" void kernel_launch(void* const* d_in, const int* in_sizes, int n_in,
                              void* d_out, int out_size, void* d_ws, size_t ws_size,
                              hipStream_t stream)
{
  (void)in_sizes; (void)n_in; (void)out_size; (void)ws_size;
  const int*   sent  = (const int*)  d_in[0];
  const float* emb   = (const float*)d_in[1];
  const float* WihF  = (const float*)d_in[2];
  const float* WhhF  = (const float*)d_in[3];
  const float* bihF  = (const float*)d_in[4];
  const float* bhhF  = (const float*)d_in[5];
  const float* WihB  = (const float*)d_in[6];
  const float* WhhB  = (const float*)d_in[7];
  const float* bihB  = (const float*)d_in[8];
  const float* bhhB  = (const float*)d_in[9];
  const float* Wfc   = (const float*)d_in[10];
  const float* bfc   = (const float*)d_in[11];
  const float* trans = (const float*)d_in[12];

  char* ws = (char*)d_ws;
  size_t off = 0;
  auto alloc = [&](size_t bytes) {
    void* p = ws + off;
    off += (bytes + 255) & ~(size_t)255;
    return p;
  };
  unsigned* XP    = (unsigned*)alloc(sizeof(unsigned) * (size_t)T_LEN * 128);
  unsigned* WihP  = (unsigned*)alloc(sizeof(unsigned) * 2 * 128 * G4);
  unsigned* WhhTP = (unsigned*)alloc(sizeof(unsigned) * 2 * 128 * G4);
  float*    biasC = (float*)   alloc(sizeof(float)    * 2 * G4);
  float*    GX    = (float*)   alloc(sizeof(float)    * 2 * (size_t)T_LEN * G4);
  unsigned* hsP   = (unsigned*)alloc(sizeof(unsigned) * (size_t)T_LEN * 256);
  unsigned* WfcP  = (unsigned*)alloc(sizeof(unsigned) * 256 * 16);
  float*    feats = (float*)   alloc(sizeof(float)    * (size_t)T_LEN * K_TAGS);
  int*      bpbuf = (int*)     alloc(sizeof(int)      * (size_t)T_LEN * K_TAGS);

  prep_kernel<<<1024, 256, 0, stream>>>(sent, emb, WihF, WihB, WhhF, WhhB,
                                        bihF, bhhF, bihB, bhhB, Wfc,
                                        XP, WihP, WhhTP, biasC, WfcP);
  dim3 g1(128, 8, 2);
  inproj_gemm<<<g1, 256, 0, stream>>>(XP, WihP, biasC, GX);
  lstm_kernel<<<2, 1024, 0, stream>>>(WhhTP, GX, hsP);
  fc_feats<<<128, 32, 0, stream>>>(hsP, WfcP, bfc, feats);
  viterbi_kernel<<<1, 32, 0, stream>>>(feats, trans, bpbuf, (float*)d_out);
}
